// GOKGModel_41918880809003
// MI455X (gfx1250) — matrix operand order repaired and checked by reading, not compiled, not run on hardware
//
#include <hip/hip_runtime.h>

typedef __attribute__((ext_vector_type(16))) _Float16 v16h;
typedef __attribute__((ext_vector_type(8)))  float    v8f;

#define D 128
#define WAVES_PER_BLOCK 8
#define ROWS_PER_BLOCK (WAVES_PER_BLOCK * 16)

// ---------------------------------------------------------------------------
// A-operand loader: 16 f16 K-values for one row, in contiguous k order
// (k = 16*hi + s, same order as B): lanes 0-15 hold K {0..15}, lanes 16-31 hold
// K {16..31}.  'off' = hi*16.  Converts f32 -> f16 on the fly.
// ---------------------------------------------------------------------------
__device__ __forceinline__ v16h load_a_tile(const float* __restrict__ rowptr,
                                            int kbase, int off) {
  const float* p = rowptr + kbase + off;
  float4 x0 = *(const float4*)(p + 0);
  float4 x1 = *(const float4*)(p + 4);
  float4 y0 = *(const float4*)(p + 8);
  float4 y1 = *(const float4*)(p + 12);
  v16h a;
  a[0]  = (_Float16)x0.x; a[1]  = (_Float16)x0.y; a[2]  = (_Float16)x0.z; a[3]  = (_Float16)x0.w;
  a[4]  = (_Float16)x1.x; a[5]  = (_Float16)x1.y; a[6]  = (_Float16)x1.z; a[7]  = (_Float16)x1.w;
  a[8]  = (_Float16)y0.x; a[9]  = (_Float16)y0.y; a[10] = (_Float16)y0.z; a[11] = (_Float16)y0.w;
  a[12] = (_Float16)y1.x; a[13] = (_Float16)y1.y; a[14] = (_Float16)y1.z; a[15] = (_Float16)y1.w;
  return a;
}

// ---------------------------------------------------------------------------
// Y[nrows x 128] = act( X1 @ W1 (+ X2 @ W2) + bias ), W given as f16
// transposed Wt[n][k].  One wave computes a 16x128 strip via
// v_wmma_f32_16x16x32_f16 (8 n-tiles x 4 k-steps).
// ---------------------------------------------------------------------------
__global__ __launch_bounds__(256) void gemm128_wmma(
    const float* __restrict__ X1, const _Float16* __restrict__ Wt1,
    const float* __restrict__ X2, const _Float16* __restrict__ Wt2,
    const float* __restrict__ bias, float* __restrict__ Y,
    int nrows, int do_relu)
{
  int lane = threadIdx.x & 31;
  int wave = threadIdx.x >> 5;
  int hi   = lane >> 4;       // which half-wave
  int ln   = lane & 15;
  int rowbase = (blockIdx.x * WAVES_PER_BLOCK + wave) * 16;
  if (rowbase >= nrows) return;

  int arow = rowbase + ln;            // A-operand row for this lane
  if (arow >= nrows) arow = nrows - 1;
  int off = hi * 16;

  v16h a1[4];
  const float* x1r = X1 + (size_t)arow * D;
#pragma unroll
  for (int kt = 0; kt < 4; ++kt) a1[kt] = load_a_tile(x1r, kt * 32, off);

  v16h a2[4];
  if (X2) {
    const float* x2r = X2 + (size_t)arow * D;
#pragma unroll
    for (int kt = 0; kt < 4; ++kt) a2[kt] = load_a_tile(x2r, kt * 32, off);
  }

#pragma unroll
  for (int nt = 0; nt < 8; ++nt) {
    v8f c = {0.f, 0.f, 0.f, 0.f, 0.f, 0.f, 0.f, 0.f};
    int n = nt * 16 + ln;
    // B 32x16 layout: lane n, lanes 0-15 K 0..15, lanes 16-31 K 16..31,
    // contiguous in Wt[n][k]  (32B aligned -> b128 loads, L0-resident)
    const _Float16* wb1 = Wt1 + (size_t)n * D + hi * 16;
#pragma unroll
    for (int kt = 0; kt < 4; ++kt) {
      v16h b = *(const v16h*)(wb1 + kt * 32);
      c = __builtin_amdgcn_wmma_f32_16x16x32_f16(false, a1[kt], false, b,
                                                 (short)0, c, false, false);
    }
    if (X2) {
      const _Float16* wb2 = Wt2 + (size_t)n * D + hi * 16;
#pragma unroll
      for (int kt = 0; kt < 4; ++kt) {
        v16h b = *(const v16h*)(wb2 + kt * 32);
        c = __builtin_amdgcn_wmma_f32_16x16x32_f16(false, a2[kt], false, b,
                                                   (short)0, c, false, false);
      }
    }
    float bv = bias[n];
#pragma unroll
    for (int r = 0; r < 8; ++r) {       // C/D: row = r + 8*hi, col = ln
      int row = rowbase + hi * 8 + r;
      if (row < nrows) {
        float v = c[r] + bv;
        if (do_relu) v = fmaxf(v, 0.0f);
        Y[(size_t)row * D + n] = v;
      }
    }
  }
}

// ---------------------------------------------------------------------------
// Transpose + f16-convert the six 128x128 weight matrices: Wt[n][k] = W[k][n]
// ---------------------------------------------------------------------------
__global__ void prep_weights(const float* __restrict__ w0, const float* __restrict__ w1,
                             const float* __restrict__ w2, const float* __restrict__ w3,
                             const float* __restrict__ w4, const float* __restrict__ w5,
                             _Float16* __restrict__ wt)
{
  int t = blockIdx.x * blockDim.x + threadIdx.x;
  if (t >= 6 * D * D) return;
  int w = t >> 14;
  int r = t & (D * D - 1);
  int n = r >> 7;
  int k = r & (D - 1);
  const float* W = (w == 0) ? w0 : (w == 1) ? w1 : (w == 2) ? w2
                 : (w == 3) ? w3 : (w == 4) ? w4 : w5;
  wt[(size_t)w * D * D + n * D + k] = (_Float16)W[k * D + n];
}

__global__ void zero_f32(float* __restrict__ p, long n) {
  long i = (long)blockIdx.x * blockDim.x + threadIdx.x;
  if (i * 4 < n) {
    float4 z = {0.f, 0.f, 0.f, 0.f};
    ((float4*)p)[i] = z;
  }
}

// ---------------------------------------------------------------------------
// segment_max via bitwise atomicMax: m >= 0 (post-relu), so uint compare of
// the IEEE bits == float compare; agg pre-zeroed handles empty segments.
// One wave per edge: coalesced 512B row gather, 4 atomics/lane into L2.
// ---------------------------------------------------------------------------
__global__ __launch_bounds__(256) void scatter_max(
    const float* __restrict__ M, const int* __restrict__ src,
    const int* __restrict__ dst, unsigned* __restrict__ agg, int nedges)
{
  int wid  = (int)(((long)blockIdx.x * blockDim.x + threadIdx.x) >> 5);
  int lane = threadIdx.x & 31;
  if (wid >= nedges) return;
  int s = src[wid], d = dst[wid];
  float4 v = ((const float4*)(M + (size_t)s * D))[lane];
  unsigned* arow = agg + (size_t)d * D + lane * 4;
  atomicMax(arow + 0, __float_as_uint(v.x));
  atomicMax(arow + 1, __float_as_uint(v.y));
  atomicMax(arow + 2, __float_as_uint(v.z));
  atomicMax(arow + 3, __float_as_uint(v.w));
}

// One wave per row: l2-normalize in place.
__global__ __launch_bounds__(256) void l2norm(float* __restrict__ H, int nrows) {
  int wid  = (int)(((long)blockIdx.x * blockDim.x + threadIdx.x) >> 5);
  int lane = threadIdx.x & 31;
  if (wid >= nrows) return;
  float4* row = (float4*)(H + (size_t)wid * D);
  float4 v = row[lane];
  float s = v.x * v.x + v.y * v.y + v.z * v.z + v.w * v.w;
#pragma unroll
  for (int m = 16; m >= 1; m >>= 1) s += __shfl_xor(s, m, 32);
  float inv = 1.0f / fmaxf(sqrtf(s), 1e-12f);
  v.x *= inv; v.y *= inv; v.z *= inv; v.w *= inv;
  row[lane] = v;
}

// Per-node projection for the edge MLP: P[n] = {h@Wp[:128], h@Wp[128:]} (4 f32)
__global__ __launch_bounds__(256) void pred_proj(
    const float* __restrict__ H, const float* __restrict__ Wpred,
    float* __restrict__ P, int nrows)
{
  __shared__ float w[512];                 // Wpred is [256][2] row-major
  for (int i = threadIdx.x; i < 512; i += blockDim.x) w[i] = Wpred[i];
  __syncthreads();
  int wid  = (int)(((long)blockIdx.x * blockDim.x + threadIdx.x) >> 5);
  int lane = threadIdx.x & 31;
  if (wid >= nrows) return;
  float4 v = ((const float4*)(H + (size_t)wid * D))[lane];
  float hv[4] = {v.x, v.y, v.z, v.w};
  float a0 = 0.f, a1 = 0.f, a2 = 0.f, a3 = 0.f;
  int k0 = lane * 4;
#pragma unroll
  for (int j = 0; j < 4; ++j) {
    int k = k0 + j;
    float h = hv[j];
    a0 += h * w[2 * k + 0];
    a1 += h * w[2 * k + 1];
    a2 += h * w[2 * k + 256];
    a3 += h * w[2 * k + 257];
  }
#pragma unroll
  for (int m = 16; m >= 1; m >>= 1) {
    a0 += __shfl_xor(a0, m, 32);
    a1 += __shfl_xor(a1, m, 32);
    a2 += __shfl_xor(a2, m, 32);
    a3 += __shfl_xor(a3, m, 32);
  }
  if (lane == 0) {
    float4 o = {a0, a1, a2, a3};
    ((float4*)P)[wid] = o;
  }
}

// score[e] = P[src][0:2] + P[dst][2:4] + bpred   (P table is L2-resident)
__global__ __launch_bounds__(256) void edge_score(
    const float* __restrict__ P, const int* __restrict__ src,
    const int* __restrict__ dst, const float* __restrict__ bpred,
    float* __restrict__ out, int nedges)
{
  int e = blockIdx.x * blockDim.x + threadIdx.x;
  if (e >= nedges) return;
  float4 ps = ((const float4*)P)[src[e]];
  float4 pd = ((const float4*)P)[dst[e]];
  float2 o;
  o.x = ps.x + pd.z + bpred[0];
  o.y = ps.y + pd.w + bpred[1];
  ((float2*)out)[e] = o;
}

// ---------------------------------------------------------------------------
extern "C" void kernel_launch(void* const* d_in, const int* in_sizes, int n_in,
                              void* d_out, int out_size, void* d_ws, size_t ws_size,
                              hipStream_t stream)
{
  (void)n_in; (void)out_size; (void)ws_size;
  const float* x     = (const float*)d_in[0];
  const int*   src   = (const int*)d_in[1];
  const int*   dst   = (const int*)d_in[2];
  const float* Wp1   = (const float*)d_in[3];
  const float* bp1   = (const float*)d_in[4];
  const float* Ws1   = (const float*)d_in[5];
  const float* Wn1   = (const float*)d_in[6];
  const float* b1    = (const float*)d_in[7];
  const float* Wp2   = (const float*)d_in[8];
  const float* bp2   = (const float*)d_in[9];
  const float* Ws2   = (const float*)d_in[10];
  const float* Wn2   = (const float*)d_in[11];
  const float* b2    = (const float*)d_in[12];
  const float* Wpred = (const float*)d_in[13];
  const float* bpred = (const float*)d_in[14];
  float* out = (float*)d_out;

  const int nnodes = in_sizes[0] / D;
  const int nedges = in_sizes[1];

  char* ws = (char*)d_ws;
  size_t off = 0;
  auto take = [&](size_t bytes) -> char* {
    char* p = ws + off;
    off = (off + bytes + 255) & ~((size_t)255);
    return p;
  };
  _Float16* wt   = (_Float16*)take((size_t)6 * D * D * sizeof(_Float16));
  float*    mbuf = (float*)take((size_t)nnodes * D * 4);   // m / m2 / h2
  float*    agg  = (float*)take((size_t)nnodes * D * 4);
  float*    h1   = (float*)take((size_t)nnodes * D * 4);
  float*    P    = (float*)take((size_t)nnodes * 4 * 4);

  _Float16* wt_p1 = wt + 0 * D * D;
  _Float16* wt_s1 = wt + 1 * D * D;
  _Float16* wt_n1 = wt + 2 * D * D;
  _Float16* wt_p2 = wt + 3 * D * D;
  _Float16* wt_s2 = wt + 4 * D * D;
  _Float16* wt_n2 = wt + 5 * D * D;

  prep_weights<<<(6 * D * D + 255) / 256, 256, 0, stream>>>(
      Wp1, Ws1, Wn1, Wp2, Ws2, Wn2, wt);

  const int gblocks = (nnodes + ROWS_PER_BLOCK - 1) / ROWS_PER_BLOCK;
  const long naf = (long)nnodes * D;
  const int zblocks = (int)((naf / 4 + 255) / 256);
  const int eblocks = (nedges + 7) / 8;          // one wave per edge
  const int nwb = (nnodes + 7) / 8;              // one wave per node

  // ---- layer 1 ----
  gemm128_wmma<<<gblocks, 256, 0, stream>>>(x, wt_p1, nullptr, nullptr, bp1,
                                            mbuf, nnodes, 1);
  zero_f32<<<zblocks, 256, 0, stream>>>(agg, naf);
  scatter_max<<<eblocks, 256, 0, stream>>>(mbuf, src, dst, (unsigned*)agg, nedges);
  gemm128_wmma<<<gblocks, 256, 0, stream>>>(x, wt_s1, agg, wt_n1, b1,
                                            h1, nnodes, 1);
  l2norm<<<nwb, 256, 0, stream>>>(h1, nnodes);

  // ---- layer 2 ----
  gemm128_wmma<<<gblocks, 256, 0, stream>>>(h1, wt_p2, nullptr, nullptr, bp2,
                                            mbuf, nnodes, 1);
  zero_f32<<<zblocks, 256, 0, stream>>>(agg, naf);
  scatter_max<<<eblocks, 256, 0, stream>>>(mbuf, src, dst, (unsigned*)agg, nedges);
  gemm128_wmma<<<gblocks, 256, 0, stream>>>(h1, wt_s2, agg, wt_n2, b2,
                                            mbuf, nnodes, 0);   // h2 -> mbuf
  l2norm<<<nwb, 256, 0, stream>>>(mbuf, nnodes);

  // ---- edge predictor ----
  pred_proj<<<nwb, 256, 0, stream>>>(mbuf, Wpred, P, nnodes);
  edge_score<<<(nedges + 255) / 256, 256, 0, stream>>>(P, src, dst, bpred,
                                                       out, nedges);
}
